// DynamicMoELayer_15539191677318
// MI455X (gfx1250) — matrix operand order repaired and checked by reading, not compiled, not run on hardware
//
#include <hip/hip_runtime.h>
#include <hip/hip_bf16.h>
#include <math.h>

#define T_TOK 8192
#define DIM   1024
#define DFFN  4096
#define NEXP  8

typedef __attribute__((ext_vector_type(16))) __bf16 v16bf;
typedef __attribute__((ext_vector_type(8)))  __bf16 v8bf;
typedef __attribute__((ext_vector_type(8)))  float  v8f;

#define TM     32            // tokens per block (2 x 16-row WMMA tiles)
#define NCHUNK 128           // DFF chunk per fused iteration
#define XSTR   (DIM + 8)     // LDS row stride (bf16), 16B aligned, bank-skewed
#define HSTR   (NCHUNK + 8)

// ---------------------------------------------------------------- zero output
__global__ void zero_out_kernel(float* __restrict__ out) {
    size_t i = ((size_t)blockIdx.x * 256 + threadIdx.x) * 4;
    if (i < (size_t)T_TOK * DIM) {
        float4 z = {0.f, 0.f, 0.f, 0.f};
        *(float4*)(out + i) = z;
    }
}

// ---------------------------------------------------------------- f32 -> bf16 weight staging
__global__ void cvt_bf16_kernel(const float* __restrict__ src, __bf16* __restrict__ dst) {
    size_t i = ((size_t)blockIdx.x * 256 + threadIdx.x) * 8;
    float4 a = *(const float4*)(src + i);
    float4 b = *(const float4*)(src + i + 4);
    v8bf o;
    o[0] = (__bf16)a.x; o[1] = (__bf16)a.y; o[2] = (__bf16)a.z; o[3] = (__bf16)a.w;
    o[4] = (__bf16)b.x; o[5] = (__bf16)b.y; o[6] = (__bf16)b.z; o[7] = (__bf16)b.w;
    *(v8bf*)(dst + i) = o;
}

// ---------------------------------------------------------------- gating: one wave per token
__global__ void __launch_bounds__(256)
gating_kernel(const float* __restrict__ x, const float* __restrict__ gw,
              int* __restrict__ top_i, float* __restrict__ top_w,
              float* __restrict__ probs, int* __restrict__ argm) {
    int wave = threadIdx.x >> 5;
    int lane = threadIdx.x & 31;
    int t = blockIdx.x * 8 + wave;
    if (t >= T_TOK) return;

    float part[NEXP];
#pragma unroll
    for (int e = 0; e < NEXP; e++) part[e] = 0.f;

    const float* xr = x + (size_t)t * DIM;
    for (int d = lane; d < DIM; d += 32) {
        float xv = xr[d];
#pragma unroll
        for (int e = 0; e < NEXP; e++) part[e] += xv * gw[e * DIM + d];
    }
#pragma unroll
    for (int e = 0; e < NEXP; e++) {
#pragma unroll
        for (int off = 16; off > 0; off >>= 1)
            part[e] += __shfl_xor(part[e], off, 32);
    }

    if (lane == 0) {
        float v0 = part[0]; int a0 = 0;
#pragma unroll
        for (int e = 1; e < NEXP; e++) if (part[e] > v0) { v0 = part[e]; a0 = e; }
        float v1 = -3.4e38f; int a1 = 0;
#pragma unroll
        for (int e = 0; e < NEXP; e++) if (e != a0 && part[e] > v1) { v1 = part[e]; a1 = e; }
        float e1 = expf(v1 - v0);
        float g0 = 1.0f / (1.0f + e1);
        float g1 = e1 / (1.0f + e1);
        top_i[t * 2 + 0] = a0;  top_w[t * 2 + 0] = g0;
        top_i[t * 2 + 1] = a1;  top_w[t * 2 + 1] = g1;
        argm[t] = a0;
        float s = 0.f;
        float ex[NEXP];
#pragma unroll
        for (int e = 0; e < NEXP; e++) { ex[e] = expf(part[e] - v0); s += ex[e]; }
        float inv = 1.0f / s;
#pragma unroll
        for (int e = 0; e < NEXP; e++) probs[(size_t)t * NEXP + e] = ex[e] * inv;
    }
}

// ---------------------------------------------------------------- deterministic per-expert compaction
__global__ void __launch_bounds__(256)
build_lists_kernel(const int* __restrict__ top_i, const float* __restrict__ top_w,
                   int* __restrict__ list_tok, float* __restrict__ list_w,
                   int* __restrict__ counts) {
    int e = blockIdx.x;
    int tid = threadIdx.x;
    __shared__ int sc[256];
    __shared__ int s_off;
    if (tid == 0) s_off = 0;
    __syncthreads();

    for (int base = 0; base < T_TOK; base += 256) {
        int t = base + tid;
        int m = 0; float g = 0.f;
        int i0 = top_i[t * 2], i1 = top_i[t * 2 + 1];
        if (i0 == e)      { m = 1; g = top_w[t * 2]; }
        else if (i1 == e) { m = 1; g = top_w[t * 2 + 1]; }

        sc[tid] = m;
        __syncthreads();
        for (int d = 1; d < 256; d <<= 1) {
            int v = sc[tid];
            if (tid >= d) v += sc[tid - d];
            __syncthreads();
            sc[tid] = v;
            __syncthreads();
        }
        int pos = s_off + sc[tid] - m;
        if (m) {
            list_tok[(size_t)e * T_TOK + pos] = t;
            list_w[(size_t)e * T_TOK + pos]   = g;
        }
        __syncthreads();
        if (tid == 255) s_off += sc[255];
        __syncthreads();
    }
    if (tid == 0) counts[e] = s_off;
}

// ---------------------------------------------------------------- deterministic stats reduce
__global__ void __launch_bounds__(256)
reduce_stats_kernel(const float* __restrict__ probs, const int* __restrict__ argm,
                    float* __restrict__ sumP, float* __restrict__ cntA) {
    int e = blockIdx.x, tid = threadIdx.x;
    __shared__ float sp[256], sa[256];
    float p = 0.f, a = 0.f;
    for (int t = tid; t < T_TOK; t += 256) {
        p += probs[(size_t)t * NEXP + e];
        a += (argm[t] == e) ? 1.f : 0.f;
    }
    sp[tid] = p; sa[tid] = a;
    __syncthreads();
    for (int s = 128; s > 0; s >>= 1) {
        if (tid < s) { sp[tid] += sp[tid + s]; sa[tid] += sa[tid + s]; }
        __syncthreads();
    }
    if (tid == 0) { sumP[e] = sp[0]; cntA[e] = sa[0]; }
}

__global__ void lb_kernel(const float* __restrict__ sumP, const float* __restrict__ cntA,
                          float* __restrict__ out_lb) {
    if (threadIdx.x == 0) {
        float s = 0.f;
        const float invT = 1.0f / (float)T_TOK;
#pragma unroll
        for (int e = 0; e < NEXP; e++) s += (cntA[e] * invT) * (sumP[e] * invT);
        *out_lb = (float)NEXP * s;
    }
}

// ---------------------------------------------------------------- fused MoE FFN: bf16 WMMA
// STAGED=true: weights pre-converted to bf16 in workspace (no cvt in hot loop).
template <bool STAGED>
__global__ void __launch_bounds__(256)
moe_ffn_kernel(const float* __restrict__ x,
               const float* __restrict__ w1f, const __bf16* __restrict__ w1b,
               const float* __restrict__ b1,
               const float* __restrict__ w2f, const __bf16* __restrict__ w2b,
               const float* __restrict__ b2,
               const int* __restrict__ counts,
               const int* __restrict__ list_tok, const float* __restrict__ list_w,
               float* __restrict__ out) {
    __shared__ __align__(16) __bf16 x_lds[TM * XSTR];     // 66 KB
    __shared__ __align__(16) __bf16 h_lds[TM * HSTR];     // 8.7 KB
    __shared__ int   s_tok[TM];
    __shared__ float s_gw[TM];

    const int e    = blockIdx.x >> 8;        // expert
    const int tile = blockIdx.x & 255;       // 32-token tile within expert list
    const int cnt  = counts[e];
    if (tile * TM >= cnt) return;

    const int tid   = threadIdx.x;
    const int wave  = tid >> 5;
    const int lane  = tid & 31;
    const int nlane = lane & 15;             // N index (B/C/D striping)
    const int mrow  = lane & 15;             // M row (A fragment)
    const int koffA = (lane >> 4) << 4;      // A: hi half-wave holds K+16 block
    const int koffB = (lane >> 4) << 4;      // B: hi half-wave holds K+16 block
    const int rbase = (lane >> 4) << 3;      // C/D: hi half-wave holds M+8 rows

    if (tid < TM) {
        int idx = tile * TM + tid;
        int tok = 0; float g = 0.f;
        if (idx < cnt) {
            tok = list_tok[(size_t)e * T_TOK + idx];
            g   = list_w[(size_t)e * T_TOK + idx];
        }
        s_tok[tid] = tok; s_gw[tid] = g;     // padded rows: token 0, weight 0
    }
    __syncthreads();

    // Gather x tile (TM x 1024) into LDS as bf16, float4-vectorized.
    for (int i = tid * 4; i < TM * DIM; i += 256 * 4) {
        int r = i >> 10, c = i & (DIM - 1);
        float4 v = *(const float4*)(x + (size_t)s_tok[r] * DIM + c);
        __bf16* p = &x_lds[r * XSTR + c];
        p[0] = (__bf16)v.x; p[1] = (__bf16)v.y; p[2] = (__bf16)v.z; p[3] = (__bf16)v.w;
    }
    __syncthreads();

    // A fragment (16x32 bf16) from LDS: elem i -> k = i + koffA
    auto loadA = [&](const __bf16* base, int kbase) -> v16bf {
        const __bf16* p = base + mrow * ((base == h_lds || base == h_lds + 16 * HSTR) ? HSTR : XSTR)
                          + kbase + koffA;
        v8bf lo = *(const v8bf*)(p);
        v8bf hi = *(const v8bf*)(p + 8);
        v16bf a;
#pragma unroll
        for (int i = 0; i < 8; i++) { a[i] = lo[i]; a[8 + i] = hi[i]; }
        return a;
    };
    // B fragment (32x16): lane n holds 16 consecutive k starting at koffB.
    auto loadB_f32 = [&](const float* rowbase, int kbase) -> v16bf {
        const float* p = rowbase + kbase + koffB;
        v16bf b;
#pragma unroll
        for (int i = 0; i < 16; i += 4) {
            float4 f = *(const float4*)(p + i);
            b[i] = (__bf16)f.x; b[i + 1] = (__bf16)f.y; b[i + 2] = (__bf16)f.z; b[i + 3] = (__bf16)f.w;
        }
        return b;
    };
    auto loadB_bf = [&](const __bf16* rowbase, int kbase) -> v16bf {
        const __bf16* p = rowbase + kbase + koffB;
        v8bf lo = *(const v8bf*)(p);
        v8bf hi = *(const v8bf*)(p + 8);
        v16bf b;
#pragma unroll
        for (int i = 0; i < 8; i++) { b[i] = lo[i]; b[8 + i] = hi[i]; }
        return b;
    };

    v8f acc[2][8];                            // y: wave owns D cols [wave*128, +128), 2 M tiles
#pragma unroll
    for (int m = 0; m < 2; m++)
#pragma unroll
        for (int i = 0; i < 8; i++) acc[m][i] = v8f{};

    for (int fc = 0; fc < DFFN; fc += NCHUNK) {
        // ---- layer 1: h[TM][128]; wave owns h cols [wave*16, +16)
        {
            const int ncol = fc + wave * 16 + nlane;                 // w1 row (dff index)
            const float*  w1rf = w1f + ((size_t)e * DFFN + ncol) * DIM;
            const __bf16* w1rb = w1b + ((size_t)e * DFFN + ncol) * DIM;
            v8f h0 = v8f{}, h1 = v8f{};
            for (int kb = 0; kb < DIM; kb += 32) {
                v16bf b;
                if constexpr (STAGED) b = loadB_bf(w1rb, kb);
                else                  b = loadB_f32(w1rf, kb);
                v16bf a0 = loadA(x_lds, kb);
                v16bf a1 = loadA(x_lds + 16 * XSTR, kb);
                h0 = __builtin_amdgcn_wmma_f32_16x16x32_bf16(false, a0, false, b, (short)0, h0, false, false);
                h1 = __builtin_amdgcn_wmma_f32_16x16x32_bf16(false, a1, false, b, (short)0, h1, false, false);
            }
            float bias = b1[e * DFFN + fc + wave * 16 + nlane];
#pragma unroll
            for (int r = 0; r < 8; r++) {
                float v = h0[r] + bias;
                v = 0.5f * v * (1.0f + erff(v * 0.70710678118654752f));   // exact GELU
                h_lds[(rbase + r) * HSTR + wave * 16 + nlane] = (__bf16)v;
                float u = h1[r] + bias;
                u = 0.5f * u * (1.0f + erff(u * 0.70710678118654752f));
                h_lds[(16 + rbase + r) * HSTR + wave * 16 + nlane] = (__bf16)u;
            }
        }
        __syncthreads();
        // ---- layer 2: y[TM][wave slab] += h @ w2_chunk^T
        for (int kb = 0; kb < NCHUNK; kb += 32) {
            v16bf a0 = loadA(h_lds, kb);
            v16bf a1 = loadA(h_lds + 16 * HSTR, kb);
#pragma unroll
            for (int nt = 0; nt < 8; nt++) {
                const int ncol = wave * 128 + nt * 16 + nlane;       // output D col = w2 row
                v16bf b;
                if constexpr (STAGED)
                    b = loadB_bf(w2b + ((size_t)e * DIM + ncol) * DFFN + fc, kb);
                else
                    b = loadB_f32(w2f + ((size_t)e * DIM + ncol) * DFFN + fc, kb);
                acc[0][nt] = __builtin_amdgcn_wmma_f32_16x16x32_bf16(false, a0, false, b, (short)0, acc[0][nt], false, false);
                acc[1][nt] = __builtin_amdgcn_wmma_f32_16x16x32_bf16(false, a1, false, b, (short)0, acc[1][nt], false, false);
            }
        }
        __syncthreads();   // h_lds reused next chunk
    }

    // Epilogue: +b2, gate scale, scatter-add (2 contributions/token -> deterministic)
#pragma unroll
    for (int mt = 0; mt < 2; mt++) {
#pragma unroll
        for (int nt = 0; nt < 8; nt++) {
            int dcol = wave * 128 + nt * 16 + nlane;
            float bias = b2[e * DIM + dcol];
#pragma unroll
            for (int r = 0; r < 8; r++) {
                int row = mt * 16 + rbase + r;
                float y = acc[mt][nt][r] + bias;
                atomicAdd(&out[(size_t)s_tok[row] * DIM + dcol], s_gw[row] * y);
            }
        }
    }
}

// ---------------------------------------------------------------- launcher
extern "C" void kernel_launch(void* const* d_in, const int* in_sizes, int n_in,
                              void* d_out, int out_size, void* d_ws, size_t ws_size,
                              hipStream_t stream) {
    const float* x   = (const float*)d_in[0];
    const float* gw  = (const float*)d_in[1];
    const float* w1  = (const float*)d_in[2];
    const float* b1  = (const float*)d_in[3];
    const float* w2  = (const float*)d_in[4];
    const float* b2  = (const float*)d_in[5];
    float* out = (float*)d_out;

    // ---- workspace carve-up: small arrays first, then optional bf16 weight staging
    char* base = (char*)d_ws;
    size_t off = 0;
    auto carve = [&](size_t bytes) { void* p = base + off; off = (off + bytes + 255) & ~(size_t)255; return p; };
    int*   counts   = (int*)  carve(NEXP * sizeof(int));
    float* sumP     = (float*)carve(NEXP * sizeof(float));
    float* cntA     = (float*)carve(NEXP * sizeof(float));
    int*   top_i    = (int*)  carve((size_t)T_TOK * 2 * sizeof(int));
    float* top_w    = (float*)carve((size_t)T_TOK * 2 * sizeof(float));
    int*   argm     = (int*)  carve((size_t)T_TOK * sizeof(int));
    float* probs    = (float*)carve((size_t)T_TOK * NEXP * sizeof(float));
    int*   list_tok = (int*)  carve((size_t)NEXP * T_TOK * sizeof(int));
    float* list_w   = (float*)carve((size_t)NEXP * T_TOK * sizeof(float));
    const size_t NW = (size_t)NEXP * DFFN * DIM;          // elements in w1 (== w2)
    __bf16* w1b = (__bf16*)carve(NW * sizeof(__bf16));
    __bf16* w2b = (__bf16*)carve(NW * sizeof(__bf16));
    const bool staged = (off <= ws_size);                  // fixed per process -> deterministic

    // 1) zero output
    zero_out_kernel<<<(T_TOK * DIM) / (256 * 4), 256, 0, stream>>>(out);
    // 2) optional bf16 weight staging (removes cvt from hot loop, halves L2 weight bytes)
    if (staged) {
        cvt_bf16_kernel<<<NW / (256 * 8), 256, 0, stream>>>(w1, w1b);
        cvt_bf16_kernel<<<NW / (256 * 8), 256, 0, stream>>>(w2, w2b);
    }
    // 3) gating
    gating_kernel<<<T_TOK / 8, 256, 0, stream>>>(x, gw, top_i, top_w, probs, argm);
    // 4) deterministic per-expert token compaction
    build_lists_kernel<<<NEXP, 256, 0, stream>>>(top_i, top_w, list_tok, list_w, counts);
    // 5) load-balance loss -> d_out[T*D]
    reduce_stats_kernel<<<NEXP, 256, 0, stream>>>(probs, argm, sumP, cntA);
    lb_kernel<<<1, 32, 0, stream>>>(sumP, cntA, out + (size_t)T_TOK * DIM);
    // 6) fused WMMA expert FFN with gated scatter-add
    if (staged)
        moe_ffn_kernel<true><<<NEXP * 256, 256, 0, stream>>>(x, w1, w1b, b1, w2, w2b, b2,
                                                             counts, list_tok, list_w, out);
    else
        moe_ffn_kernel<false><<<NEXP * 256, 256, 0, stream>>>(x, w1, w1b, b1, w2, w2b, b2,
                                                              counts, list_tok, list_w, out);
}
